// GTS_Forecasting_Module_59691455480208
// MI455X (gfx1250) — matrix operand order repaired and checked by reading, not compiled, not run on hardware
//
#include <hip/hip_runtime.h>
#include <hip/hip_bf16.h>
#include <type_traits>

#define N_NODES 10000
#define NP      10112      // padded node count: multiple of 128 (M tiles) and 32 (K chunks)
#define HID     64
#define EMB     64
#define T_ENC   12
#define T_DEC   12
#define SPLITK  4
#define BPAD    48         // LDS row pitch (elements); 96B rows keep 32B alignment
// First NT_ROW_START adjacency rows are kept L2-resident (7168*10112*2 = 145 MB < 192 MB L2);
// rows beyond are streamed with non-temporal loads so they don't evict the resident set.
#define NT_ROW_START (56 * 128)

typedef __bf16 bf16_t;
typedef __attribute__((ext_vector_type(16))) __bf16 v16bf;
typedef __attribute__((ext_vector_type(8)))  __bf16 v8bf;
typedef __attribute__((ext_vector_type(8)))  float  v8f;

__device__ __forceinline__ bf16_t f2bf(float f) { return (bf16_t)f; }

// ---------------------------------------------------------------------------
// C(M x NT*16, f32) = A(M x K, bf16 row-major) @ B(K x NT*16) given as
// Bt (NT*16 x K, bf16 row-major, i.e. column-major B). Split-K partials via
// blockIdx.y -> C + ks*partStride. 256 threads = 8 waves; wave w owns rows
// [m0, m0+16) x all NT column tiles. Double-buffered LDS staging for B; the
// RT/NT A-load decision is hoisted out of the K loop; last iteration peeled.
// ---------------------------------------------------------------------------
template<int NT>
__global__ __launch_bounds__(256) void gemm_bf16_wmma(
    const bf16_t* __restrict__ A, int lda,
    const bf16_t* __restrict__ Bt, int ldb,
    float* __restrict__ C, int ldc,
    int Kseg, long partStride, int ntRowStart)
{
  __shared__ __align__(64) bf16_t ldsB[2][128 * BPAD];

  const int tid  = threadIdx.x;
  const int wave = tid >> 5;
  const int lane = tid & 31;
  const int m0   = blockIdx.x * 128 + wave * 16;
  const int k0   = blockIdx.y * Kseg;
  C += (long)blockIdx.y * partStride;

  // uniform per block: stream rows >= ntRowStart with TH=NT (preserve L2-resident set)
  const bool aNT = (int)(blockIdx.x * 128) >= ntRowStart;

  v8f acc[NT] = {};

  // A fragment addressing (contiguous k order, same as B fragment):
  // lane<16: row=lane,   K = 0..15;  lane>=16: row=lane-16, K = 16..31
  const int arow = m0 + (lane & 15);
  const int kbA  = (lane < 16) ? 0 : 16;
  // B fragment: lane<16: col=lane, K = 0..15; lane>=16: col=lane-16, K = 16..31
  const int kbB  = (lane < 16) ? 0 : 16;
  const int nr   = lane & 15;

  // LDS staging map: NT*16 cols x 32 k-elems, NT*2 bf16 per thread, contiguous
  constexpr int ELEM = NT * 2;
  const int fl  = tid * ELEM;
  const int bn  = fl >> 5;
  const int bkk = fl & 31;

  const bf16_t* aRow = A + (long)arow * lda + k0 + kbA;
  const bf16_t* bCol = Bt + (long)bn * ldb + k0 + bkk;

  auto stage = [&](int kk, int buf) {
    bf16_t* dst = &ldsB[buf][bn * BPAD + bkk];
    if constexpr (NT == 8) *(v16bf*)dst = *(const v16bf*)(bCol + kk);
    else                   *(v8bf*)dst  = *(const v8bf*)(bCol + kk);
  };

  auto compute = [&](int kk, int cur, auto antc) {
    constexpr bool ANT = decltype(antc)::value;
    v8bf alo, ahi;
    if constexpr (ANT) {
      alo = __builtin_nontemporal_load((const v8bf*)(aRow + kk));
      ahi = __builtin_nontemporal_load((const v8bf*)(aRow + kk + 8));
    } else {
      alo = *(const v8bf*)(aRow + kk);
      ahi = *(const v8bf*)(aRow + kk + 8);
      __builtin_prefetch(aRow + kk + 512, 0, 1);   // global_prefetch_b8 on resident A stream
    }
    v16bf afrag = __builtin_shufflevector(alo, ahi,
        0,1,2,3,4,5,6,7,8,9,10,11,12,13,14,15);
#pragma unroll
    for (int j = 0; j < NT; ++j) {
      v16bf bfrag = *(const v16bf*)&ldsB[cur][(j * 16 + nr) * BPAD + kbB];
      acc[j] = __builtin_amdgcn_wmma_f32_16x16x32_bf16(
          false, afrag, false, bfrag, (short)0, acc[j], false, false);
    }
  };

  stage(0, 0);
  __syncthreads();

  auto mainloop = [&](auto antc) {
    int k = 0;
    for (; k < Kseg - 32; k += 32) {          // staging unconditional; tail peeled
      const int cur = (k >> 5) & 1;
      stage(k + 32, cur ^ 1);
      compute(k, cur, antc);
      __syncthreads();
    }
    compute(k, (k >> 5) & 1, antc);
  };
  if (aNT) mainloop(std::integral_constant<bool, true>{});
  else     mainloop(std::integral_constant<bool, false>{});

  // C/D f32 layout: VGPR r, lane<16 -> row m0+r, col lane; lane>=16 -> row m0+8+r
  const int rbase = m0 + ((lane >> 4) << 3);
  const int cc    = lane & 15;
#pragma unroll
  for (int j = 0; j < NT; ++j)
#pragma unroll
    for (int r = 0; r < 8; ++r)
      C[(long)(rbase + r) * ldc + j * 16 + cc] = acc[j][r];
}

// ---------------------------------------------------------------------------
// Sum `nparts` split-K partials of S (NP x 128 f32, pitch partStride), write
// bf16 transpose T (128 x ldT), and write the f32 sum back into partial 0.
// block (32,8), grid (NP/32, 4)
// ---------------------------------------------------------------------------
__global__ __launch_bounds__(256) void transpose_sum_to_bf16(
    float* __restrict__ S, long partStride, int nparts,
    bf16_t* __restrict__ T, int ldT)
{
  __shared__ float tile[32][33];
  const int tx = threadIdx.x, ty = threadIdx.y;
  const int i0 = blockIdx.x * 32, d0 = blockIdx.y * 32;
#pragma unroll
  for (int s = 0; s < 32; s += 8) {
    long idx = (long)(i0 + ty + s) * 128 + d0 + tx;
    float v = S[idx];
    for (int p = 1; p < nparts; ++p) v += S[idx + (long)p * partStride];
    tile[ty + s][tx] = v;
    if (nparts > 1) S[idx] = v;
  }
  __syncthreads();
#pragma unroll
  for (int s = 0; s < 32; s += 8)
    T[(long)(d0 + ty + s) * ldT + i0 + tx] = f2bf(tile[tx][ty + s]);
}

// ---- elementwise / prep kernels -------------------------------------------

__global__ void prep_adjb(const float* __restrict__ adj, bf16_t* __restrict__ adjb) {
  int k = blockIdx.x * 256 + threadIdx.x;
  int i = blockIdx.y;
  if (k >= NP) return;
  float v = (i < N_NODES && k < N_NODES) ? adj[(long)i * N_NODES + k] : 0.f;
  adjb[(long)i * NP + k] = f2bf(v);
}

__global__ void prep_wt_enc(const float* __restrict__ W, bf16_t* __restrict__ Wt, int nout) {
  int idx = blockIdx.x * 256 + threadIdx.x;
  if (idx >= nout * 384) return;
  int n = idx / 384, kk = idx % 384;
  Wt[idx] = f2bf(W[(long)kk * nout + n]);
}

// decoder weights: rows padded 65 -> 128 per concat block
__global__ void prep_wt_dec(const float* __restrict__ W, bf16_t* __restrict__ Wt, int nout) {
  int idx = blockIdx.x * 256 + threadIdx.x;
  if (idx >= nout * 384) return;
  int n = idx / 384, kk = idx % 384;
  int b = kk >> 7, j = kk & 127;
  float v = (j < 65) ? W[(long)(b * 65 + j) * nout + n] : 0.f;
  Wt[idx] = f2bf(v);
}

__global__ void zero_f32(float* __restrict__ p, long n) {
  long idx = (long)blockIdx.x * 256 + threadIdx.x;
  if (idx < n) p[idx] = 0.f;
}

// cat = [embed(x_t), (r?)*h]  (NP x 128)
__global__ void build_cat_enc(const float* __restrict__ inputs, int t,
                              const float* __restrict__ W_emb, const float* __restrict__ b_emb,
                              const float* __restrict__ h, const float* __restrict__ ru,
                              float* __restrict__ cat)
{
  long idx = (long)blockIdx.x * 256 + threadIdx.x;
  if (idx >= (long)NP * 128) return;
  int i = (int)(idx >> 7), d = (int)(idx & 127);
  float v = 0.f;
  if (i < N_NODES) {
    if (d < EMB) {
      v = inputs[(long)i * T_ENC + t] * W_emb[d] + b_emb[d];
    } else {
      float hv = h[(long)i * HID + (d - EMB)];
      if (ru) hv *= ru[(long)i * 128 + (d - EMB)];
      v = hv;
    }
  }
  cat[idx] = v;
}

// cat = [x(1), (r?)*h(64), 0...]  (NP x 128)
__global__ void build_cat_dec(const float* __restrict__ pred,
                              const float* __restrict__ h, const float* __restrict__ ru,
                              float* __restrict__ cat)
{
  long idx = (long)blockIdx.x * 256 + threadIdx.x;
  if (idx >= (long)NP * 128) return;
  int i = (int)(idx >> 7), d = (int)(idx & 127);
  float v = 0.f;
  if (i < N_NODES) {
    if (d == 0) {
      v = pred[i];
    } else if (d <= HID) {
      float hv = h[(long)i * HID + (d - 1)];
      if (ru) hv *= ru[(long)i * 128 + (d - 1)];
      v = hv;
    }
  }
  cat[idx] = v;
}

// x2 = 2 * sum_partials(Y) - cat
__global__ void x2_combine(const float* __restrict__ Y, long partStride,
                           const float* __restrict__ cat, float* __restrict__ x2)
{
  long idx = (long)blockIdx.x * 256 + threadIdx.x;
  if (idx >= (long)NP * 128) return;
  float s = Y[idx];
#pragma unroll
  for (int p = 1; p < SPLITK; ++p) s += Y[idx + (long)p * partStride];
  x2[idx] = 2.f * s - cat[idx];
}

// concat3 bf16 (NP x 384) = [cat | x1 | x2]
__global__ void build_c3(const float* __restrict__ cat, const float* __restrict__ x1,
                         const float* __restrict__ x2, bf16_t* __restrict__ c3)
{
  long idx = (long)blockIdx.x * 256 + threadIdx.x;
  if (idx >= (long)NP * 384) return;
  int i = (int)(idx / 384), c = (int)(idx % 384);
  const float* src = (c < 128) ? cat : ((c < 256) ? x1 : x2);
  c3[idx] = f2bf(src[(long)i * 128 + (c & 127)]);
}

__global__ void sigmoid_bias(const float* __restrict__ g, const float* __restrict__ bg,
                             float* __restrict__ ru)
{
  long idx = (long)blockIdx.x * 256 + threadIdx.x;
  if (idx >= (long)NP * 128) return;
  float z = g[idx] + bg[idx & 127];
  ru[idx] = 1.f / (1.f + __expf(-z));
}

// h = u*h + (1-u)*tanh(gc + bc)
__global__ void h_update(const float* __restrict__ gc, const float* __restrict__ bc,
                         const float* __restrict__ ru, float* __restrict__ h)
{
  long idx = (long)blockIdx.x * 256 + threadIdx.x;
  if (idx >= (long)NP * HID) return;
  int i = (int)(idx >> 6), d = (int)(idx & 63);
  float c = tanhf(gc[idx] + bc[d]);
  float u = ru[(long)i * 128 + HID + d];
  h[idx] = u * h[idx] + (1.f - u) * c;
}

__global__ void predict(const float* __restrict__ h, const float* __restrict__ W_pred,
                        const float* __restrict__ b_pred, float* __restrict__ predbuf,
                        float* __restrict__ out, int t)
{
  int i = blockIdx.x * 256 + threadIdx.x;
  if (i >= NP) return;
  float p = 0.f;
  if (i < N_NODES) {
    p = b_pred[0];
#pragma unroll
    for (int j = 0; j < HID; ++j) p += h[(long)i * HID + j] * W_pred[j];
    out[(long)i * T_DEC + t] = p;
  }
  predbuf[i] = p;
}

// ---------------------------------------------------------------------------

extern "C" void kernel_launch(void* const* d_in, const int* in_sizes, int n_in,
                              void* d_out, int out_size, void* d_ws, size_t ws_size,
                              hipStream_t stream)
{
  (void)in_sizes; (void)n_in; (void)out_size; (void)ws_size;
  const float* inputs = (const float*)d_in[0];
  const float* adj    = (const float*)d_in[2];
  const float* W_emb  = (const float*)d_in[3];
  const float* b_emb  = (const float*)d_in[4];
  const float* Wg_e   = (const float*)d_in[5];
  const float* bg_e   = (const float*)d_in[6];
  const float* Wc_e   = (const float*)d_in[7];
  const float* bc_e   = (const float*)d_in[8];
  const float* Wg_d   = (const float*)d_in[9];
  const float* bg_d   = (const float*)d_in[10];
  const float* Wc_d   = (const float*)d_in[11];
  const float* bc_d   = (const float*)d_in[12];
  const float* W_pred = (const float*)d_in[13];
  const float* b_pred = (const float*)d_in[14];
  float* outp = (float*)d_out;

  size_t off = 0;
  auto carve = [&](size_t bytes) -> void* {
    void* p = (char*)d_ws + off;
    off += (bytes + 255) & ~(size_t)255;
    return p;
  };
  bf16_t* adjb = (bf16_t*)carve((size_t)NP * NP * 2);
  float*  cat  = (float*) carve((size_t)NP * 128 * 4);
  bf16_t* catT = (bf16_t*)carve((size_t)NP * 128 * 2);
  float*  x1p  = (float*) carve((size_t)SPLITK * NP * 128 * 4);
  bf16_t* x1T  = (bf16_t*)carve((size_t)NP * 128 * 2);
  float*  yp   = (float*) carve((size_t)SPLITK * NP * 128 * 4);
  float*  x2   = (float*) carve((size_t)NP * 128 * 4);
  bf16_t* c3b  = (bf16_t*)carve((size_t)NP * 384 * 2);
  float*  g    = (float*) carve((size_t)NP * 128 * 4);
  float*  ru   = (float*) carve((size_t)NP * 128 * 4);
  float*  gc   = (float*) carve((size_t)NP * 64 * 4);
  float*  hbuf = (float*) carve((size_t)NP * 64 * 4);
  float*  pred = (float*) carve((size_t)NP * 4);
  bf16_t* WgTe = (bf16_t*)carve((size_t)128 * 384 * 2);
  bf16_t* WcTe = (bf16_t*)carve((size_t)64 * 384 * 2);
  bf16_t* WgTd = (bf16_t*)carve((size_t)128 * 384 * 2);
  bf16_t* WcTd = (bf16_t*)carve((size_t)64 * 384 * 2);

  // one-time preps (re-run each call; deterministic)
  prep_adjb<<<dim3((NP + 255) / 256, NP), 256, 0, stream>>>(adj, adjb);
  prep_wt_enc<<<(128 * 384 + 255) / 256, 256, 0, stream>>>(Wg_e, WgTe, 128);
  prep_wt_enc<<<(64 * 384 + 255) / 256, 256, 0, stream>>>(Wc_e, WcTe, 64);
  prep_wt_dec<<<(128 * 384 + 255) / 256, 256, 0, stream>>>(Wg_d, WgTd, 128);
  prep_wt_dec<<<(64 * 384 + 255) / 256, 256, 0, stream>>>(Wc_d, WcTd, 64);
  zero_f32<<<(int)(((long)NP * 64 + 255) / 256), 256, 0, stream>>>(hbuf, (long)NP * 64);
  zero_f32<<<(NP + 255) / 256, 256, 0, stream>>>(pred, NP);

  const long ew  = (long)NP * 128;
  const int  ewb = (int)((ew + 255) / 256);
  const int  c3blk = (int)(((long)NP * 384 + 255) / 256);
  const int  hblk  = (int)(((long)NP * 64 + 255) / 256);
  dim3 tgrid(NP / 32, 4), tblk(32, 8);
  dim3 gBig(NP / 128, SPLITK), gSmall(NP / 128, 1);
  const long pstr = (long)NP * 128;
  const int  ALL_RT = 0x40000000;   // small GEMMs: everything regular-temporal

  // x1 = adj@cat ; x2 = 2*adj@x1 - cat ; c3b = bf16([cat|x1|x2])
  auto diffusion = [&]() {
    transpose_sum_to_bf16<<<tgrid, tblk, 0, stream>>>(cat, 0, 1, catT, NP);
    gemm_bf16_wmma<8><<<gBig, 256, 0, stream>>>(adjb, NP, catT, NP, x1p, 128,
                                                NP / SPLITK, pstr, NT_ROW_START);
    transpose_sum_to_bf16<<<tgrid, tblk, 0, stream>>>(x1p, pstr, SPLITK, x1T, NP);
    gemm_bf16_wmma<8><<<gBig, 256, 0, stream>>>(adjb, NP, x1T, NP, yp, 128,
                                                NP / SPLITK, pstr, NT_ROW_START);
    x2_combine<<<ewb, 256, 0, stream>>>(yp, pstr, cat, x2);
    build_c3<<<c3blk, 256, 0, stream>>>(cat, x1p, x2, c3b);
  };

  for (int t = 0; t < T_ENC; ++t) {
    build_cat_enc<<<ewb, 256, 0, stream>>>(inputs, t, W_emb, b_emb, hbuf, nullptr, cat);
    diffusion();
    gemm_bf16_wmma<8><<<gSmall, 256, 0, stream>>>(c3b, 384, WgTe, 384, g, 128, 384, 0, ALL_RT);
    sigmoid_bias<<<ewb, 256, 0, stream>>>(g, bg_e, ru);
    build_cat_enc<<<ewb, 256, 0, stream>>>(inputs, t, W_emb, b_emb, hbuf, ru, cat);
    diffusion();
    gemm_bf16_wmma<4><<<gSmall, 256, 0, stream>>>(c3b, 384, WcTe, 384, gc, 64, 384, 0, ALL_RT);
    h_update<<<hblk, 256, 0, stream>>>(gc, bc_e, ru, hbuf);
  }

  for (int t = 0; t < T_DEC; ++t) {
    build_cat_dec<<<ewb, 256, 0, stream>>>(pred, hbuf, nullptr, cat);
    diffusion();
    gemm_bf16_wmma<8><<<gSmall, 256, 0, stream>>>(c3b, 384, WgTd, 384, g, 128, 384, 0, ALL_RT);
    sigmoid_bias<<<ewb, 256, 0, stream>>>(g, bg_d, ru);
    build_cat_dec<<<ewb, 256, 0, stream>>>(pred, hbuf, ru, cat);
    diffusion();
    gemm_bf16_wmma<4><<<gSmall, 256, 0, stream>>>(c3b, 384, WcTd, 384, gc, 64, 384, 0, ALL_RT);
    h_update<<<hblk, 256, 0, stream>>>(gc, bc_d, ru, hbuf);
    predict<<<(NP + 255) / 256, 256, 0, stream>>>(hbuf, W_pred, b_pred, pred, outp, t);
  }
}
